// CrossGraphAttention_9010841387640
// MI455X (gfx1250) — matrix operand order repaired and checked by reading, not compiled, not run on hardware
//
#include <hip/hip_runtime.h>

// ---------------------------------------------------------------------------
// CrossGraphAttention for MI455X (gfx1250), wave32, WMMA bf16 16x16x32.
//
// Passes (all on `stream`, flash-style, distance matrix never materialized):
//   0) zero stats               (rowmax/rowsum/colmax/colsum)
//   1) prep                     h_i,h_j -> bf16 copies (+ h_j^T), row norms
//   2) stats_max                tile-WMMA S, a = sqrt(...), atomic max
//   3) stats_sum                tile-WMMA S, exp(a-max), atomicAdd sums
//   4) update_i                 h_i_new = h_i - a_i @ h_j   (+ h_i_new^T bf16)
//   5) update_j                 h_j_new = h_j - a_j^T @ h_i_new
//
// All GEMMs are in NT form so both WMMA operands load as two contiguous
// 16-byte chunks per lane (ISA §7.12.2 fragment layouts).  Softmax epilogue
// uses raw TRANS ops (v_sqrt_f32 / v_exp_f32 / v_rcp_f32) so it co-executes
// with the matrix pipe instead of expanding into long VALU cndmask chains.
// ---------------------------------------------------------------------------

typedef __attribute__((ext_vector_type(16))) __bf16       v16bf;
typedef __attribute__((ext_vector_type(8)))  float        v8f;
typedef __attribute__((ext_vector_type(4)))  unsigned int v4u;

#define DEV static __device__ __forceinline__

constexpr int kN = 16384;
constexpr int kM = 16384;
constexpr int kD = 128;

struct FragBF { union { v16bf v; unsigned int u[8]; }; };

DEV v8f v8f_zero() {
  v8f z = {0.f, 0.f, 0.f, 0.f, 0.f, 0.f, 0.f, 0.f};
  return z;
}

// fast hardware TRANS ops (good to ~1 ulp; inputs here are well-scaled)
DEV float fast_sqrt(float x) { return __builtin_amdgcn_sqrtf(x); }
DEV float fast_rcp(float x)  { return __builtin_amdgcn_rcpf(x); }

// A-fragment (16x32 bf16, M=lane&15): K chunks at +8*(lane>=16) and +16.
DEV void loadA(FragBF& f, const __bf16* base, int ld) {
  const int lane = threadIdx.x & 31;
  const __bf16* p = base + (size_t)(lane & 15) * ld + ((lane >> 4) << 3);
  v4u a = *(const v4u*)p;
  v4u b = *(const v4u*)(p + 16);
  f.u[0] = a[0]; f.u[1] = a[1]; f.u[2] = a[2]; f.u[3] = a[3];
  f.u[4] = b[0]; f.u[5] = b[1]; f.u[6] = b[2]; f.u[7] = b[3];
}

// B-fragment (32x16 bf16, N=lane&15): K chunks at +8*(lane>=16) and +16.
DEV void loadB(FragBF& f, const __bf16* base, int ld) {
  const int lane = threadIdx.x & 31;
  const __bf16* p = base + (size_t)(lane & 15) * ld + ((lane >> 4) << 3);
  v4u a = *(const v4u*)p;
  v4u b = *(const v4u*)(p + 16);
  f.u[0] = a[0]; f.u[1] = a[1]; f.u[2] = a[2]; f.u[3] = a[3];
  f.u[4] = b[0]; f.u[5] = b[1]; f.u[6] = b[2]; f.u[7] = b[3];
}

DEV v8f wmma_bf16(const FragBF& a, const FragBF& b, v8f c) {
  return __builtin_amdgcn_wmma_f32_16x16x32_bf16(
      /*neg_a=*/false, a.v, /*neg_b=*/false, b.v,
      /*c_mod=*/(short)0, c, /*reuse_a=*/false, /*reuse_b=*/false);
}

// distances are >= 0, so uint compare == float compare
DEV void atomicMaxPos(float* addr, float v) {
  atomicMax((unsigned int*)addr, __float_as_uint(v));
}

// ---------------------------------------------------------------------------
__global__ void k_zero(float* p, int n) {
  int i = blockIdx.x * blockDim.x + threadIdx.x;
  if (i < n) p[i] = 0.f;
}

// one block (128 thr) per row: bf16 copies, h_j^T, squared norms
__global__ void k_prep(const float* __restrict__ h_i, const float* __restrict__ h_j,
                       __bf16* __restrict__ hi_bf, __bf16* __restrict__ hj_bf,
                       __bf16* __restrict__ hjT, float* __restrict__ ni2,
                       float* __restrict__ nj2) {
  __shared__ float red[128];
  const int r = blockIdx.x, t = threadIdx.x;
  float x;
  if (r < kN) {
    x = h_i[(size_t)r * kD + t];
    hi_bf[(size_t)r * kD + t] = (__bf16)x;
  } else {
    const int rr = r - kN;
    x = h_j[(size_t)rr * kD + t];
    __bf16 b = (__bf16)x;
    hj_bf[(size_t)rr * kD + t] = b;
    hjT[(size_t)t * kM + rr] = b;
  }
  red[t] = x * x;
  __syncthreads();
  for (int s = 64; s > 0; s >>= 1) {
    if (t < s) red[t] += red[t + s];
    __syncthreads();
  }
  if (t == 0) { if (r < kN) ni2[r] = red[0]; else nj2[r - kN] = red[0]; }
}

// ---------------------------------------------------------------------------
// Stats kernels: grid (kM/64, kN/64), 256 thr = 8 waves.
// wave w: i-tile (w&3), j-pair (w>>2) covering 2 column tiles of 16.
__global__ void k_stats_max(const __bf16* __restrict__ hi_bf,
                            const __bf16* __restrict__ hj_bf,
                            const float* __restrict__ ni2,
                            const float* __restrict__ nj2,
                            float* __restrict__ rowmax, float* __restrict__ colmax) {
  const int lane = threadIdx.x & 31, w = threadIdx.x >> 5;
  const int ib = blockIdx.y * 64 + (w & 3) * 16;
  const int jb = blockIdx.x * 64 + (w >> 2) * 32;

  FragBF A[4];
#pragma unroll
  for (int kk = 0; kk < 4; ++kk) loadA(A[kk], hi_bf + (size_t)ib * kD + kk * 32, kD);

  float niv[8];
#pragma unroll
  for (int r = 0; r < 8; ++r) niv[r] = ni2[ib + r + ((lane >> 4) << 3)];

  float rowm[8];
#pragma unroll
  for (int r = 0; r < 8; ++r) rowm[r] = 0.f;

#pragma unroll
  for (int jt = 0; jt < 2; ++jt) {
    const int j0 = jb + jt * 16;
    v8f s = v8f_zero();
    FragBF B;
#pragma unroll
    for (int kk = 0; kk < 4; ++kk) {
      loadB(B, hj_bf + (size_t)j0 * kD + kk * 32, kD);
      s = wmma_bf16(A[kk], B, s);
    }
    const float njv = nj2[j0 + (lane & 15)];
    float cm = 0.f;
#pragma unroll
    for (int r = 0; r < 8; ++r) {
      float d2 = niv[r] + njv - 2.f * s[r];
      float a = fast_sqrt(fmaxf(d2, 0.f));
      cm = fmaxf(cm, a);
      rowm[r] = fmaxf(rowm[r], a);
    }
    cm = fmaxf(cm, __shfl_xor(cm, 16));
    if (lane < 16) atomicMaxPos(&colmax[j0 + lane], cm);
  }
#pragma unroll
  for (int r = 0; r < 8; ++r) {
    rowm[r] = fmaxf(rowm[r], __shfl_xor(rowm[r], 1));
    rowm[r] = fmaxf(rowm[r], __shfl_xor(rowm[r], 2));
    rowm[r] = fmaxf(rowm[r], __shfl_xor(rowm[r], 4));
    rowm[r] = fmaxf(rowm[r], __shfl_xor(rowm[r], 8));
  }
  if ((lane & 15) == 0) {
    const int i0 = ib + ((lane >> 4) << 3);
#pragma unroll
    for (int r = 0; r < 8; ++r) atomicMaxPos(&rowmax[i0 + r], rowm[r]);
  }
}

__global__ void k_stats_sum(const __bf16* __restrict__ hi_bf,
                            const __bf16* __restrict__ hj_bf,
                            const float* __restrict__ ni2,
                            const float* __restrict__ nj2,
                            const float* __restrict__ rowmax,
                            const float* __restrict__ colmax,
                            float* __restrict__ rowsum, float* __restrict__ colsum) {
  const int lane = threadIdx.x & 31, w = threadIdx.x >> 5;
  const int ib = blockIdx.y * 64 + (w & 3) * 16;
  const int jb = blockIdx.x * 64 + (w >> 2) * 32;

  FragBF A[4];
#pragma unroll
  for (int kk = 0; kk < 4; ++kk) loadA(A[kk], hi_bf + (size_t)ib * kD + kk * 32, kD);

  float niv[8], rmv[8], rs[8];
#pragma unroll
  for (int r = 0; r < 8; ++r) {
    const int ig = ib + r + ((lane >> 4) << 3);
    niv[r] = ni2[ig];
    rmv[r] = rowmax[ig];
    rs[r] = 0.f;
  }

#pragma unroll
  for (int jt = 0; jt < 2; ++jt) {
    const int j0 = jb + jt * 16;
    v8f s = v8f_zero();
    FragBF B;
#pragma unroll
    for (int kk = 0; kk < 4; ++kk) {
      loadB(B, hj_bf + (size_t)j0 * kD + kk * 32, kD);
      s = wmma_bf16(A[kk], B, s);
    }
    const float njv = nj2[j0 + (lane & 15)];
    const float cmx = colmax[j0 + (lane & 15)];
    float cs = 0.f;
#pragma unroll
    for (int r = 0; r < 8; ++r) {
      float d2 = niv[r] + njv - 2.f * s[r];
      float a = fast_sqrt(fmaxf(d2, 0.f));
      cs += __expf(a - cmx);       // column softmax numerator (over i)
      rs[r] += __expf(a - rmv[r]); // row softmax numerator (over j)
    }
    cs += __shfl_xor(cs, 16);
    if (lane < 16) atomicAdd(&colsum[j0 + lane], cs);
  }
#pragma unroll
  for (int r = 0; r < 8; ++r) {
    rs[r] += __shfl_xor(rs[r], 1);
    rs[r] += __shfl_xor(rs[r], 2);
    rs[r] += __shfl_xor(rs[r], 4);
    rs[r] += __shfl_xor(rs[r], 8);
  }
  if ((lane & 15) == 0) {
    const int i0 = ib + ((lane >> 4) << 3);
#pragma unroll
    for (int r = 0; r < 8; ++r) atomicAdd(&rowsum[i0 + r], rs[r]);
  }
}

// ---------------------------------------------------------------------------
// h_i_new = h_i - a_i @ h_j.  grid kN/64, 256 thr = 8 waves.
// wave w: i-tile (w&3), d-group (w>>2) covering 64 of the 128 feature cols.
__global__ void k_update_i(const float* __restrict__ h_i,
                           const __bf16* __restrict__ hi_bf,
                           const __bf16* __restrict__ hj_bf,
                           const __bf16* __restrict__ hjT,
                           const float* __restrict__ ni2,
                           const float* __restrict__ nj2,
                           const float* __restrict__ colmax,
                           const float* __restrict__ colsum,
                           float* __restrict__ out_i, __bf16* __restrict__ hiTn) {
  __shared__ __bf16 lds[8][16][40];  // wave-private 16x32 P staging (+pad)
  const int lane = threadIdx.x & 31, w = threadIdx.x >> 5;
  const int ib = blockIdx.x * 64 + (w & 3) * 16;
  const int db = (w >> 2) * 64;

  FragBF A[4];
#pragma unroll
  for (int kk = 0; kk < 4; ++kk) loadA(A[kk], hi_bf + (size_t)ib * kD + kk * 32, kD);

  float niv[8];
#pragma unroll
  for (int r = 0; r < 8; ++r) niv[r] = ni2[ib + r + ((lane >> 4) << 3)];

  v8f acc[4];
#pragma unroll
  for (int dt = 0; dt < 4; ++dt) acc[dt] = v8f_zero();

  for (int m = 0; m < kM; m += 32) {
#pragma unroll
    for (int jt = 0; jt < 2; ++jt) {
      const int j0 = m + jt * 16;
      v8f s = v8f_zero();
      FragBF B;
#pragma unroll
      for (int kk = 0; kk < 4; ++kk) {
        loadB(B, hj_bf + (size_t)j0 * kD + kk * 32, kD);
        s = wmma_bf16(A[kk], B, s);
      }
      const int jg = j0 + (lane & 15);
      const float njv = nj2[jg];
      const float cmx = colmax[jg];
      const float rin = fast_rcp(colsum[jg]);  // colsum >= 1
#pragma unroll
      for (int r = 0; r < 8; ++r) {
        float d2 = niv[r] + njv - 2.f * s[r];
        float a = fast_sqrt(fmaxf(d2, 0.f));
        float p = __expf(a - cmx) * rin;
        lds[w][r + ((lane >> 4) << 3)][jt * 16 + (lane & 15)] = (__bf16)p;
      }
    }
    __syncthreads();
    FragBF P;
    loadA(P, &lds[w][0][0], 40);
#pragma unroll
    for (int dt = 0; dt < 4; ++dt) {
      FragBF B;
      loadB(B, hjT + (size_t)(db + dt * 16) * kM + m, kM);
      acc[dt] = wmma_bf16(P, B, acc[dt]);
    }
    __syncthreads();
    if (m + 32 < kM) __builtin_prefetch(hj_bf + (size_t)(m + 32) * kD, 0, 0);
  }

#pragma unroll
  for (int dt = 0; dt < 4; ++dt) {
    const int dg = db + dt * 16 + (lane & 15);
#pragma unroll
    for (int r = 0; r < 8; ++r) {
      const int ig = ib + r + ((lane >> 4) << 3);
      float v = h_i[(size_t)ig * kD + dg] - acc[dt][r];
      out_i[(size_t)ig * kD + dg] = v;
      hiTn[(size_t)dg * kN + ig] = (__bf16)v;  // transposed bf16 for pass 5
    }
  }
}

// h_j_new = h_j - a_j^T @ h_i_new.  Computes S^T tiles directly (NT form).
__global__ void k_update_j(const float* __restrict__ h_j,
                           const __bf16* __restrict__ hj_bf,
                           const __bf16* __restrict__ hi_bf,
                           const __bf16* __restrict__ hiTn,
                           const float* __restrict__ ni2,
                           const float* __restrict__ nj2,
                           const float* __restrict__ rowmax,
                           const float* __restrict__ rowsum,
                           float* __restrict__ out_j) {
  __shared__ __bf16 lds[8][16][40];
  const int lane = threadIdx.x & 31, w = threadIdx.x >> 5;
  const int mb = blockIdx.x * 64 + (w & 3) * 16;
  const int db = (w >> 2) * 64;

  FragBF A[4];
#pragma unroll
  for (int kk = 0; kk < 4; ++kk) loadA(A[kk], hj_bf + (size_t)mb * kD + kk * 32, kD);

  float njv[8];
#pragma unroll
  for (int r = 0; r < 8; ++r) njv[r] = nj2[mb + r + ((lane >> 4) << 3)];

  v8f acc[4];
#pragma unroll
  for (int dt = 0; dt < 4; ++dt) acc[dt] = v8f_zero();

  for (int n = 0; n < kN; n += 32) {
#pragma unroll
    for (int nt = 0; nt < 2; ++nt) {
      const int n0 = n + nt * 16;
      v8f s = v8f_zero();
      FragBF B;
#pragma unroll
      for (int kk = 0; kk < 4; ++kk) {
        loadB(B, hi_bf + (size_t)n0 * kD + kk * 32, kD);
        s = wmma_bf16(A[kk], B, s);  // s[m][n] = h_j[m] . h_i[n]
      }
      const int ng = n0 + (lane & 15);
      const float niw = ni2[ng];
      const float rmx = rowmax[ng];
      const float rin = fast_rcp(rowsum[ng]);
#pragma unroll
      for (int r = 0; r < 8; ++r) {
        float d2 = njv[r] + niw - 2.f * s[r];
        float a = fast_sqrt(fmaxf(d2, 0.f));
        float p = __expf(a - rmx) * rin;  // a_j[n][m], tile stored [m][n]
        lds[w][r + ((lane >> 4) << 3)][nt * 16 + (lane & 15)] = (__bf16)p;
      }
    }
    __syncthreads();
    FragBF P;
    loadA(P, &lds[w][0][0], 40);
#pragma unroll
    for (int dt = 0; dt < 4; ++dt) {
      FragBF B;
      loadB(B, hiTn + (size_t)(db + dt * 16) * kN + n, kN);
      acc[dt] = wmma_bf16(P, B, acc[dt]);
    }
    __syncthreads();
    if (n + 32 < kN) __builtin_prefetch(hi_bf + (size_t)(n + 32) * kD, 0, 0);
  }

#pragma unroll
  for (int dt = 0; dt < 4; ++dt) {
    const int dg = db + dt * 16 + (lane & 15);
#pragma unroll
    for (int r = 0; r < 8; ++r) {
      const int mg = mb + r + ((lane >> 4) << 3);
      out_j[(size_t)mg * kD + dg] = h_j[(size_t)mg * kD + dg] - acc[dt][r];
    }
  }
}

// ---------------------------------------------------------------------------
extern "C" void kernel_launch(void* const* d_in, const int* in_sizes, int n_in,
                              void* d_out, int out_size, void* d_ws, size_t ws_size,
                              hipStream_t stream) {
  (void)in_sizes; (void)n_in; (void)out_size; (void)ws_size;
  const float* h_i = (const float*)d_in[0];
  const float* h_j = (const float*)d_in[1];
  float* out = (float*)d_out;

  // workspace layout (~16.4 MB)
  char* ws = (char*)d_ws;
  constexpr size_t SZ_BF = (size_t)kN * kD * 2;  // 4 MiB per bf16 matrix
  __bf16* hi_bf = (__bf16*)(ws);
  __bf16* hj_bf = (__bf16*)(ws + SZ_BF);
  __bf16* hjT   = (__bf16*)(ws + 2 * SZ_BF);
  __bf16* hiTn  = (__bf16*)(ws + 3 * SZ_BF);
  float* ni2    = (float*)(ws + 4 * SZ_BF);
  float* nj2    = ni2 + kN;
  float* rowmax = nj2 + kM;
  float* rowsum = rowmax + kN;
  float* colmax = rowsum + kN;
  float* colsum = colmax + kM;

  k_zero<<<(4 * kN + 255) / 256, 256, 0, stream>>>(rowmax, 4 * kN);
  k_prep<<<kN + kM, 128, 0, stream>>>(h_i, h_j, hi_bf, hj_bf, hjT, ni2, nj2);

  dim3 gs(kM / 64, kN / 64);
  k_stats_max<<<gs, 256, 0, stream>>>(hi_bf, hj_bf, ni2, nj2, rowmax, colmax);
  k_stats_sum<<<gs, 256, 0, stream>>>(hi_bf, hj_bf, ni2, nj2, rowmax, colmax,
                                      rowsum, colsum);

  k_update_i<<<kN / 64, 256, 0, stream>>>(h_i, hi_bf, hj_bf, hjT, ni2, nj2,
                                          colmax, colsum, out, hiTn);
  k_update_j<<<kM / 64, 256, 0, stream>>>(h_j, hj_bf, hi_bf, hiTn, ni2, nj2,
                                          rowmax, rowsum, out + (size_t)kN * kD);
}
